// GAT_72533407695323
// MI455X (gfx1250) — matrix operand order repaired and checked by reading, not compiled, not run on hardware
//
#include <hip/hip_runtime.h>
#include <hip/hip_bf16.h>
#include <stdint.h>

#define NNODES 50000
#define NEDGES 800000
#define ETOT   (NEDGES + NNODES)
#define FIN    128
#define NCLS   10
#define NGRAPH 64
#define FMAX   256

typedef __attribute__((ext_vector_type(16))) _Float16 v16h;
typedef __attribute__((ext_vector_type(8)))  _Float16 v8h;
typedef __attribute__((ext_vector_type(8)))  float    v8f;

// ---- ordered-uint encoding so atomicMax(uint) == float max ----
__device__ __forceinline__ unsigned enc_f32(float f) {
  unsigned u = __float_as_uint(f);
  return (u & 0x80000000u) ? ~u : (u | 0x80000000u);
}
__device__ __forceinline__ float dec_f32(unsigned e) {
  unsigned u = (e & 0x80000000u) ? (e & 0x7fffffffu) : ~e;
  return __uint_as_float(u);
}

// ---------------- utility kernels ----------------
__global__ void zero_u32_kernel(unsigned* __restrict__ p, size_t n) {
  size_t i = (size_t)blockIdx.x * blockDim.x + threadIdx.x;
  if (i < n) p[i] = 0u;
}

__global__ void f32_to_f16_kernel(const float* __restrict__ s,
                                  _Float16* __restrict__ d, size_t n) {
  size_t i = (size_t)blockIdx.x * blockDim.x + threadIdx.x;
  if (i < n) d[i] = (_Float16)s[i];
}

// pack W [fi x fo] f32 row-major -> Wt [fo x Kpad] f16 (K-contiguous per column)
__global__ void pack_w16t_kernel(const float* __restrict__ W,
                                 _Float16* __restrict__ Wt,
                                 int fi, int fo, int Kpad) {
  int i = blockIdx.x * blockDim.x + threadIdx.x;
  if (i >= fo * Kpad) return;
  int n = i / Kpad, k = i - n * Kpad;
  float v = (k < fi) ? W[(size_t)k * fo + n] : 0.0f;
  Wt[(size_t)n * Kpad + k] = (_Float16)v;
}

// ---------------- self-loop attr: deg + sum per dst ----------------
__global__ void deg_asum_kernel(const int* __restrict__ dst,
                                const float* __restrict__ ea,
                                float* __restrict__ deg, float* __restrict__ asum) {
  int e = blockIdx.x * blockDim.x + threadIdx.x;
  if (e >= NEDGES) return;
  int d = dst[e];
  atomicAdd(&deg[d], 1.0f);
  atomicAdd(&asum[d], ea[e]);
}

__global__ void loop_attr_kernel(const float* __restrict__ deg,
                                 const float* __restrict__ asum,
                                 float* __restrict__ loopa) {
  int n = blockIdx.x * blockDim.x + threadIdx.x;
  if (n < NNODES) loopa[n] = asum[n] / fmaxf(deg[n], 1.0f);
}

// ---------------- WMMA GEMM: out[Nn x fo] = h[Nn x Kpad](f16) * Wt + bias ----
// one wave32 per 32x16 output macro-tile (two 16x16 M-tiles sharing one B
// fragment -> 2 v_wmma per B load); K-loop in steps of 32.
__global__ void wmma_gemm_kernel(const _Float16* __restrict__ h, int ldh,
                                 const _Float16* __restrict__ Wt, int Kpad,
                                 const float* __restrict__ bias,
                                 float* __restrict__ out, int Nn, int fo) {
  const int lane = threadIdx.x & 31;
  const int wave = threadIdx.x >> 5;
  const int tiles_n = fo >> 4;
  const int tiles_m = (Nn + 31) >> 5;            // 32 rows per wave
  const int tile = blockIdx.x * (blockDim.x >> 5) + wave;
  if (tile >= tiles_m * tiles_n) return;
  const int tm = tile / tiles_n;
  const int tn = tile - tm * tiles_n;

  const int half = lane >> 4;   // lane half selects K sub-range (ISA 7.12.2)
  const int l15  = lane & 15;
  const int n_g  = tn * 16 + l15;                // B/C/D column
  int m_a0 = tm * 32 + l15;                      // A rows (two tiles)
  int m_a1 = tm * 32 + 16 + l15;
  if (m_a0 >= Nn) m_a0 = Nn - 1;                 // clamp (stores are guarded)
  if (m_a1 >= Nn) m_a1 = Nn - 1;

  v8f acc0, acc1;
  {
    const float bv = bias[n_g];
#pragma unroll
    for (int r = 0; r < 8; ++r) { acc0[r] = bv; acc1[r] = bv; }
  }

  const _Float16* arow0 = h  + (size_t)m_a0 * ldh + half * 16;
  const _Float16* arow1 = h  + (size_t)m_a1 * ldh + half * 16;
  const _Float16* bcol  = Wt + (size_t)n_g * Kpad + half * 16;

  for (int k0 = 0; k0 < Kpad; k0 += 32) {
    if (k0 + 32 < Kpad) {
      __builtin_prefetch(arow0 + k0 + 32, 0, 3);   // locality=3 -> near (WGP) cache
      __builtin_prefetch(arow1 + k0 + 32, 0, 3);
    }
    v16h b = *(const v16h*)(bcol + k0);            // shared by both M-tiles
    v8h lo0 = *(const v8h*)(arow0 + k0);
    v8h hi0 = *(const v8h*)(arow0 + k0 + 8);
    v8h lo1 = *(const v8h*)(arow1 + k0);
    v8h hi1 = *(const v8h*)(arow1 + k0 + 8);
    v16h a0, a1;
#pragma unroll
    for (int i = 0; i < 8; ++i) {
      a0[i] = lo0[i]; a0[i + 8] = hi0[i];
      a1[i] = lo1[i]; a1[i + 8] = hi1[i];
    }
    acc0 = __builtin_amdgcn_wmma_f32_16x16x32_f16(false, a0, false, b,
                                                  (short)0, acc0, false, false);
    acc1 = __builtin_amdgcn_wmma_f32_16x16x32_f16(false, a1, false, b,
                                                  (short)0, acc1, false, false);
  }

#pragma unroll
  for (int r = 0; r < 8; ++r) {
    // D row layout: lanes 0-15 -> M=r, lanes 16-31 -> M=8+r
    int m_g0 = tm * 32 + half * 8 + r;
    int m_g1 = tm * 32 + 16 + half * 8 + r;
    if (m_g0 < Nn) out[(size_t)m_g0 * fo + n_g] = acc0[r];
    if (m_g1 < Nn) out[(size_t)m_g1 * fo + n_g] = acc1[r];
  }
}

// ---------------- edge phase (wave32 per edge) ----------------
__global__ void edge_logit_kernel(const int* __restrict__ src, const int* __restrict__ dst,
                                  const float* __restrict__ ea, const float* __restrict__ loopa,
                                  const float* __restrict__ xl, const float* __restrict__ xr,
                                  const float* __restrict__ We, const float* __restrict__ att,
                                  float* __restrict__ logit, unsigned* __restrict__ mxenc,
                                  int fo) {
  const int e = blockIdx.x * (blockDim.x >> 5) + (threadIdx.x >> 5);
  if (e >= ETOT) return;
  const int lane = threadIdx.x & 31;
  int s, d; float a;
  if (e < NEDGES) { s = src[e]; d = dst[e]; a = ea[e]; }
  else            { s = d = e - NEDGES;     a = loopa[e - NEDGES]; }
  float part = 0.0f;
  for (int c = lane; c < fo; c += 32) {
    float m = xl[(size_t)s * fo + c] + xr[(size_t)d * fo + c] + a * We[c];
    m = (m > 0.0f) ? m : 0.2f * m;               // leaky_relu(0.2)
    part += m * att[c];
  }
  for (int off = 16; off; off >>= 1) part += __shfl_xor(part, off, 32);
  if (lane == 0) {
    logit[e] = part;
    atomicMax(&mxenc[d], enc_f32(part));
  }
}

__global__ void edge_exp_kernel(const int* __restrict__ dst,
                                const float* __restrict__ logit,
                                const unsigned* __restrict__ mxenc,
                                float* __restrict__ exv, float* __restrict__ den) {
  int e = blockIdx.x * blockDim.x + threadIdx.x;
  if (e >= ETOT) return;
  int d = (e < NEDGES) ? dst[e] : (e - NEDGES);
  float ex = __expf(logit[e] - dec_f32(mxenc[d]));
  exv[e] = ex;
  atomicAdd(&den[d], ex);
}

__global__ void edge_accum_kernel(const int* __restrict__ src, const int* __restrict__ dst,
                                  const float* __restrict__ exv, const float* __restrict__ den,
                                  const float* __restrict__ xl, float* __restrict__ out,
                                  int fo) {
  const int e = blockIdx.x * (blockDim.x >> 5) + (threadIdx.x >> 5);
  if (e >= ETOT) return;
  const int lane = threadIdx.x & 31;
  int s, d;
  if (e < NEDGES) { s = src[e]; d = dst[e]; }
  else            { s = d = e - NEDGES; }
  const float alpha = exv[e] / den[d];
  for (int c = lane; c < fo; c += 32)
    atomicAdd(&out[(size_t)d * fo + c], alpha * xl[(size_t)s * fo + c]);
}

// out = relu(acc + b); write f32 (for pooling) and f16 (next layer, padded stride)
__global__ void bias_relu_kernel(const float* __restrict__ acc, const float* __restrict__ b,
                                 float* __restrict__ act32, _Float16* __restrict__ h16,
                                 int fo, int ldnext) {
  size_t i = (size_t)blockIdx.x * blockDim.x + threadIdx.x;
  if (i >= (size_t)NNODES * ldnext) return;
  int n = (int)(i / ldnext), c = (int)(i - (size_t)n * ldnext);
  float v = 0.0f;
  if (c < fo) {
    v = acc[(size_t)n * fo + c] + b[c];
    v = fmaxf(v, 0.0f);
    act32[(size_t)n * fo + c] = v;
  }
  h16[(size_t)n * ldnext + c] = (_Float16)v;
}

// ---------------- global mean pool (wave32 per node) ----------------
__global__ void pool_kernel(const float* __restrict__ act, const int* __restrict__ batch,
                            float* __restrict__ g, float* __restrict__ cnt, int C) {
  const int n = blockIdx.x * (blockDim.x >> 5) + (threadIdx.x >> 5);
  if (n >= NNODES) return;
  const int lane = threadIdx.x & 31;
  const int b = batch[n];
  if (lane == 0) atomicAdd(&cnt[b], 1.0f);
  for (int c = lane; c < C; c += 32)
    atomicAdd(&g[(size_t)b * C + c], act[(size_t)n * C + c]);
}

__global__ void pool_div_kernel(float* __restrict__ g, const float* __restrict__ cnt, int C) {
  int i = blockIdx.x * blockDim.x + threadIdx.x;
  if (i >= NGRAPH * C) return;
  g[i] /= fmaxf(cnt[i / C], 1.0f);
}

// ---------------- tiny FC layers ----------------
__global__ void fc_kernel(const float* __restrict__ in, const float* __restrict__ W,
                          const float* __restrict__ b, float* __restrict__ out,
                          int fi, int fo, int do_relu) {
  int i = blockIdx.x * blockDim.x + threadIdx.x;
  if (i >= NGRAPH * fo) return;
  int gi = i / fo, oc = i - gi * fo;
  float s = b[oc];
  for (int k = 0; k < fi; ++k) s += in[(size_t)gi * fi + k] * W[(size_t)k * fo + oc];
  if (do_relu) s = fmaxf(s, 0.0f);
  out[(size_t)gi * fo + oc] = s;
}

__global__ void log_softmax_kernel(const float* __restrict__ in, float* __restrict__ out) {
  const int g = blockIdx.x * (blockDim.x >> 5) + (threadIdx.x >> 5);
  if (g >= NGRAPH) return;
  const int lane = threadIdx.x & 31;
  float v = (lane < NCLS) ? in[g * NCLS + lane] : -1e30f;
  float m = v;
  for (int off = 16; off; off >>= 1) m = fmaxf(m, __shfl_xor(m, off, 32));
  float ex = (lane < NCLS) ? __expf(v - m) : 0.0f;
  float s = ex;
  for (int off = 16; off; off >>= 1) s += __shfl_xor(s, off, 32);
  if (lane < NCLS) out[g * NCLS + lane] = (v - m) - __logf(s);
}

// ---------------- host orchestration ----------------
extern "C" void kernel_launch(void* const* d_in, const int* in_sizes, int n_in,
                              void* d_out, int out_size, void* d_ws, size_t ws_size,
                              hipStream_t stream) {
  (void)in_sizes; (void)n_in; (void)out_size; (void)ws_size;
  const float* x     = (const float*)d_in[0];
  const int*   eidx  = (const int*)d_in[1];
  const float* ea    = (const float*)d_in[2];
  const int*   batch = (const int*)d_in[3];
  const int* src = eidx;
  const int* dst = eidx + NEDGES;

  // workspace carve-out
  char* w = (char*)d_ws;
  size_t off = 0;
  auto take = [&](size_t bytes) -> void* {
    void* p = w + off;
    off = (off + bytes + 255) & ~(size_t)255;
    return p;
  };
  _Float16* hA    = (_Float16*)take((size_t)NNODES * FMAX * 2);
  _Float16* hB    = (_Float16*)take((size_t)NNODES * FMAX * 2);
  float*    xl    = (float*)take((size_t)NNODES * FMAX * 4);   // also reused as act32
  float*    xr    = (float*)take((size_t)NNODES * FMAX * 4);
  float*    oacc  = (float*)take((size_t)NNODES * FMAX * 4);
  float*    logit = (float*)take((size_t)ETOT * 4);
  float*    exv   = (float*)take((size_t)ETOT * 4);
  unsigned* mxenc = (unsigned*)take((size_t)NNODES * 4);
  float*    den   = (float*)take((size_t)NNODES * 4);
  float*    deg   = (float*)take((size_t)NNODES * 4);
  float*    asum  = (float*)take((size_t)NNODES * 4);
  float*    loopa = (float*)take((size_t)NNODES * 4);
  _Float16* wl16  = (_Float16*)take((size_t)FMAX * FMAX * 2);
  _Float16* wr16  = (_Float16*)take((size_t)FMAX * FMAX * 2);
  float*    gbuf  = (float*)take((size_t)NGRAPH * FMAX * 4);
  float*    cnt   = (float*)take((size_t)NGRAPH * 4);
  float*    fbA   = (float*)take((size_t)NGRAPH * FMAX * 4);
  float*    fbB   = (float*)take((size_t)NGRAPH * FMAX * 4);

  const int BS = 256, WPB = 8;  // 8 wave32 per block
  auto gups = [](size_t n, int b) { return (unsigned)((n + b - 1) / b); };

  // x -> f16 activations
  f32_to_f16_kernel<<<gups((size_t)NNODES * FIN, BS), BS, 0, stream>>>(x, hA, (size_t)NNODES * FIN);

  // self-loop attr (fill_value='mean')
  zero_u32_kernel<<<gups(NNODES, BS), BS, 0, stream>>>((unsigned*)deg, NNODES);
  zero_u32_kernel<<<gups(NNODES, BS), BS, 0, stream>>>((unsigned*)asum, NNODES);
  deg_asum_kernel<<<gups(NEDGES, BS), BS, 0, stream>>>(dst, ea, deg, asum);
  loop_attr_kernel<<<gups(NNODES, BS), BS, 0, stream>>>(deg, asum, loopa);

  const int gdims[6] = {128, 16, 32, 64, 128, 256};
  _Float16* hcur = hA;
  _Float16* hnxt = hB;

  for (int L = 0; L < 5; ++L) {
    const int fi = gdims[L], fo = gdims[L + 1];
    const int ldin = (fi < 32) ? 32 : fi;              // K padded to 32 (zeros)
    const int ldnx = (fo < 32) ? 32 : fo;
    const float* Wl  = (const float*)d_in[4 + 7 * L + 0];
    const float* bl  = (const float*)d_in[4 + 7 * L + 1];
    const float* Wr  = (const float*)d_in[4 + 7 * L + 2];
    const float* br  = (const float*)d_in[4 + 7 * L + 3];
    const float* We  = (const float*)d_in[4 + 7 * L + 4];
    const float* att = (const float*)d_in[4 + 7 * L + 5];
    const float* bo  = (const float*)d_in[4 + 7 * L + 6];

    pack_w16t_kernel<<<gups((size_t)fo * ldin, BS), BS, 0, stream>>>(Wl, wl16, fi, fo, ldin);
    pack_w16t_kernel<<<gups((size_t)fo * ldin, BS), BS, 0, stream>>>(Wr, wr16, fi, fo, ldin);

    const int tiles = ((NNODES + 31) / 32) * (fo / 16);
    wmma_gemm_kernel<<<gups(tiles, WPB), BS, 0, stream>>>(hcur, ldin, wl16, ldin, bl, xl, NNODES, fo);
    wmma_gemm_kernel<<<gups(tiles, WPB), BS, 0, stream>>>(hcur, ldin, wr16, ldin, br, xr, NNODES, fo);

    zero_u32_kernel<<<gups(NNODES, BS), BS, 0, stream>>>(mxenc, NNODES);
    zero_u32_kernel<<<gups(NNODES, BS), BS, 0, stream>>>((unsigned*)den, NNODES);
    zero_u32_kernel<<<gups((size_t)NNODES * fo, BS), BS, 0, stream>>>((unsigned*)oacc, (size_t)NNODES * fo);

    edge_logit_kernel<<<gups(ETOT, WPB), BS, 0, stream>>>(src, dst, ea, loopa, xl, xr, We, att,
                                                          logit, mxenc, fo);
    edge_exp_kernel<<<gups(ETOT, BS), BS, 0, stream>>>(dst, logit, mxenc, exv, den);
    edge_accum_kernel<<<gups(ETOT, WPB), BS, 0, stream>>>(src, dst, exv, den, xl, oacc, fo);

    bias_relu_kernel<<<gups((size_t)NNODES * ldnx, BS), BS, 0, stream>>>(oacc, bo, xl, hnxt, fo, ldnx);

    _Float16* t = hcur; hcur = hnxt; hnxt = t;
  }

  // global mean pool (last act32 lives in xl, dense stride 256)
  zero_u32_kernel<<<gups((size_t)NGRAPH * FMAX, BS), BS, 0, stream>>>((unsigned*)gbuf, (size_t)NGRAPH * FMAX);
  zero_u32_kernel<<<gups(NGRAPH, BS), BS, 0, stream>>>((unsigned*)cnt, NGRAPH);
  pool_kernel<<<gups(NNODES, WPB), BS, 0, stream>>>(xl, batch, gbuf, cnt, FMAX);
  pool_div_kernel<<<gups((size_t)NGRAPH * FMAX, BS), BS, 0, stream>>>(gbuf, cnt, FMAX);

  // FC chain
  const int fdims[7] = {256, 128, 64, 32, 16, 8, NCLS};
  const float* fin = gbuf;
  float* fout = fbA;
  for (int L = 0; L < 6; ++L) {
    const float* W = (const float*)d_in[39 + 2 * L + 0];
    const float* b = (const float*)d_in[39 + 2 * L + 1];
    fc_kernel<<<gups((size_t)NGRAPH * fdims[L + 1], BS), BS, 0, stream>>>(
        fin, W, b, fout, fdims[L], fdims[L + 1], (L < 5) ? 1 : 0);
    fin = fout;
    fout = (fout == fbA) ? fbB : fbA;
  }

  log_softmax_kernel<<<gups(NGRAPH, WPB), BS, 0, stream>>>(fin, (float*)d_out);
}
